// KFCContrastiveLoss_11605001634112
// MI455X (gfx1250) — compile-verified, matrix operand order checked by reading, not run on hardware
//
#include <hip/hip_runtime.h>
#include <math.h>

typedef __attribute__((ext_vector_type(8)))  _Float16 v8h;
typedef __attribute__((ext_vector_type(16))) _Float16 v16h;
typedef __attribute__((ext_vector_type(8)))  float    v8f;

#define BHALF 4096          // B
#define DDIM  256           // D
#define NTOT  8192          // 2B
#define INV_BETA 12.5f
#define TILE_J 64           // B-rows staged in LDS per iteration (32 KB)
#define JSPLIT 32           // column strips per row-block
#define ROWS_PER_BLOCK 512  // 8 waves x (4 x 16 rows)
#define ATILES 4            // register-resident A tiles per wave

// ---------------------------------------------------------------------------
// Kernel 1: per-row norms, swapped-dot (for diagonal cosine), bias margin,
// and f16 normalized rows written to workspace for the WMMA kernel.
// One block per row, one element per thread (D == 256 == blockDim).
// ---------------------------------------------------------------------------
__global__ void __launch_bounds__(256)
normalize_kernel(const float* __restrict__ x1, const float* __restrict__ x2,
                 const float* __restrict__ bias, _Float16* __restrict__ xnh,
                 float* __restrict__ norms, float* __restrict__ dotab,
                 float* __restrict__ dm) {
  int row = blockIdx.x;
  int t   = threadIdx.x;
  const float* arow = (row < BHALF) ? (x1 + (size_t)row * DDIM)
                                    : (x2 + (size_t)(row - BHALF) * DDIM);
  const float* brow = (row < BHALF) ? (x2 + (size_t)row * DDIM)
                                    : (x1 + (size_t)(row - BHALF) * DDIM);
  float a = arow[t];
  float b = brow[t];

  __shared__ float r1[256], r2[256];
  r1[t] = a * a;
  r2[t] = a * b;
  __syncthreads();
  for (int s = 128; s > 0; s >>= 1) {
    if (t < s) { r1[t] += r1[t + s]; r2[t] += r2[t + s]; }
    __syncthreads();
  }
  float nrm = fmaxf(sqrtf(r1[0]), 1e-8f);
  xnh[(size_t)row * DDIM + t] = (_Float16)(a / nrm);
  if (t == 0) {
    norms[row] = nrm;
    dotab[row] = r2[0];
    float s4 = bias[row * 4 + 0] + bias[row * 4 + 1] +
               bias[row * 4 + 2] + bias[row * 4 + 3];
    dm[row] = s4 / (float)NTOT;
  }
}

// ---------------------------------------------------------------------------
// WMMA layout helper (wave32, 16x16x32 f16).
// A: lane -> M, halves -> K split {0..7|16..23} low lanes / {8..15|24..31}
// high lanes.  B: identical pattern with lane -> N.
// ---------------------------------------------------------------------------
__device__ __forceinline__ v16h load_tile_chunk(const _Float16* rowp, int c,
                                                int ko) {
  v8h lo = *(const v8h*)(rowp + 32 * c + ko);
  v8h hi = *(const v8h*)(rowp + 32 * c + ko + 16);
  v16h r;
#pragma unroll
  for (int h = 0; h < 8; ++h) { r[h] = lo[h]; r[8 + h] = hi[h]; }
  return r;
}

// ---------------------------------------------------------------------------
// Kernel 2: fused Gram-matrix + exp + masked row-sum via WMMA f16->f32.
// grid = (NTOT/512) row-blocks * JSPLIT column strips.
// Each wave32 owns FOUR 16-row A tiles resident in registers (rows
// ra+{0,128,256,384}), so every B chunk read from LDS feeds four WMMAs,
// dropping LDS read demand well below the bank ceiling and leaving the
// matrix pipe as the sole limiter. B tiles of 64 rows staged through LDS,
// shared by all 8 waves.
// ---------------------------------------------------------------------------
__global__ void __launch_bounds__(256)
rowsumexp_kernel(const _Float16* __restrict__ xnh, float* __restrict__ rs) {
  __shared__ __align__(16) _Float16 Bs[TILE_J * DDIM];  // 32 KB

  int rb    = blockIdx.x / JSPLIT;
  int strip = blockIdx.x % JSPLIT;
  int i0    = rb * ROWS_PER_BLOCK;
  int wave  = threadIdx.x >> 5;
  int lane  = threadIdx.x & 31;

  int mIdx = lane & 15;              // A: lane -> matrix row M
  int ko   = (lane < 16) ? 0 : 8;    // K offset of the lane half-group
  int nIdx = lane & 15;              // C/D: lane -> N
  int mOff = (lane >> 4) << 3;       // C/D: high lanes hold M = r+8

  int ra[ATILES];
#pragma unroll
  for (int u = 0; u < ATILES; ++u) ra[u] = i0 + wave * 16 + u * 128;

  // Preload the four A tiles: 4 x (16 rows x 256 K), 8 chunks of 16x32 each.
  v16h A[ATILES][8];
#pragma unroll
  for (int u = 0; u < ATILES; ++u) {
    const _Float16* arow = xnh + (size_t)(ra[u] + mIdx) * DDIM;
#pragma unroll
    for (int c = 0; c < 8; ++c) A[u][c] = load_tile_chunk(arow, c, ko);
  }

  v8f acc[ATILES];  // per-lane partial exp sums (N-reduction deferred)
#pragma unroll
  for (int u = 0; u < ATILES; ++u) acc[u] = (v8f){};

  int jstart = strip * (NTOT / JSPLIT);
  int jend   = jstart + (NTOT / JSPLIT);
  for (int j0 = jstart; j0 < jend; j0 += TILE_J) {
    __syncthreads();
    // Cooperative stage of 64 rows (32 KB) into LDS: 8 x b128 per thread.
    const _Float16* src = xnh + (size_t)j0 * DDIM;
#pragma unroll
    for (int s = 0; s < 8; ++s) {
      int idx = (s * 256 + threadIdx.x) * 8;
      *(v8h*)(Bs + idx) = *(const v8h*)(src + idx);
    }
    __syncthreads();

#pragma unroll 1
    for (int js = 0; js < TILE_J; js += 16) {
      const _Float16* brow = Bs + (size_t)(js + nIdx) * DDIM;
      v8f cs[ATILES];
#pragma unroll
      for (int u = 0; u < ATILES; ++u) cs[u] = (v8f){};
#pragma unroll
      for (int c = 0; c < 8; ++c) {
        v16h Bm = load_tile_chunk(brow, c, ko);  // one B load ...
#pragma unroll
        for (int u = 0; u < ATILES; ++u) {       // ... four WMMAs
          cs[u] = __builtin_amdgcn_wmma_f32_16x16x32_f16(
              false, A[u][c], false, Bm, (short)0, cs[u], false, false);
        }
      }
      // exp(cos/beta) with diagonal zeroed (tiles are 16-aligned, so the
      // diagonal can only appear when the tile bases coincide; then M==N).
      int jbase = j0 + js;
#pragma unroll
      for (int u = 0; u < ATILES; ++u) {
        bool dt = (jbase == ra[u]);
#pragma unroll
        for (int r = 0; r < 8; ++r) {
          float e = __expf(cs[u][r] * INV_BETA);
          if (dt && (r + mOff) == nIdx) e = 0.f;
          acc[u][r] += e;
        }
      }
    }
  }

  // Reduce over N: sum lanes 0..15 (M = r) and lanes 16..31 (M = r+8).
  float* out = rs + (size_t)strip * NTOT;
#pragma unroll
  for (int u = 0; u < ATILES; ++u) {
#pragma unroll
    for (int r = 0; r < 8; ++r) {
      float v = acc[u][r];
      v += __shfl_xor(v, 1, 32);
      v += __shfl_xor(v, 2, 32);
      v += __shfl_xor(v, 4, 32);
      v += __shfl_xor(v, 8, 32);
      if (lane == 0)       out[ra[u] + r]     = v;
      else if (lane == 16) out[ra[u] + 8 + r] = v;
    }
  }
}

// ---------------------------------------------------------------------------
// Kernel 3: deterministic final reduction -> loss (out[0]) + margins (out[1..4])
// ---------------------------------------------------------------------------
__global__ void __launch_bounds__(256)
finalize_kernel(const int* __restrict__ races, const float* __restrict__ norms,
                const float* __restrict__ dotab, const float* __restrict__ dm,
                const float* __restrict__ rs, float* __restrict__ out) {
  __shared__ float r1[256], r2[256];
  int t = threadIdx.x;

  float partial = 0.f;
  for (int i = t; i < NTOT; i += 256) {
    float rsum = 0.f;
#pragma unroll
    for (int s = 0; s < JSPLIT; ++s) rsum += rs[(size_t)s * NTOT + i];
    int isw = (i + BHALF) & (NTOT - 1);
    float diag = dotab[i] / (norms[i] * norms[isw]);
    float t1 = (diag - dm[i]) * INV_BETA;
    float den = rsum - __expf(diag * INV_BETA) + __expf(t1);
    partial += t1 - logf(den);
  }
  r1[t] = partial;
  __syncthreads();
  for (int s = 128; s > 0; s >>= 1) {
    if (t < s) r1[t] += r1[t + s];
    __syncthreads();
  }
  if (t == 0) out[0] = -r1[0] / (float)NTOT;

  // Per-race margins: pair = dm[i] + dm[i+B]; mean over 2*count.
  float ms[4] = {0.f, 0.f, 0.f, 0.f};
  float mc[4] = {0.f, 0.f, 0.f, 0.f};
  for (int i = t; i < BHALF; i += 256) {
    int r = races[i];
    float pair = dm[i] + dm[i + BHALF];
#pragma unroll
    for (int k = 0; k < 4; ++k) {
      ms[k] += (r == k) ? pair : 0.f;
      mc[k] += (r == k) ? 1.f  : 0.f;
    }
  }
  for (int k = 0; k < 4; ++k) {
    __syncthreads();
    r1[t] = ms[k];
    r2[t] = mc[k];
    __syncthreads();
    for (int s = 128; s > 0; s >>= 1) {
      if (t < s) { r1[t] += r1[t + s]; r2[t] += r2[t + s]; }
      __syncthreads();
    }
    if (t == 0) out[1 + k] = (r2[0] > 0.f) ? (r1[0] / (2.f * r2[0])) : 0.f;
  }
}

// ---------------------------------------------------------------------------
extern "C" void kernel_launch(void* const* d_in, const int* in_sizes, int n_in,
                              void* d_out, int out_size, void* d_ws,
                              size_t ws_size, hipStream_t stream) {
  (void)in_sizes; (void)n_in; (void)out_size; (void)ws_size;
  const float* x1    = (const float*)d_in[0];
  const float* x2    = (const float*)d_in[1];
  const int*   races = (const int*)d_in[2];
  const float* bias  = (const float*)d_in[3];
  float* out = (float*)d_out;

  char* ws = (char*)d_ws;
  _Float16* xnh = (_Float16*)ws;
  size_t off = (size_t)NTOT * DDIM * sizeof(_Float16);     // 4 MiB
  float* norms = (float*)(ws + off); off += (size_t)NTOT * sizeof(float);
  float* dotab = (float*)(ws + off); off += (size_t)NTOT * sizeof(float);
  float* dm    = (float*)(ws + off); off += (size_t)NTOT * sizeof(float);
  float* rs    = (float*)(ws + off); // JSPLIT * NTOT floats

  normalize_kernel<<<NTOT, 256, 0, stream>>>(x1, x2, bias, xnh, norms, dotab, dm);
  rowsumexp_kernel<<<(NTOT / ROWS_PER_BLOCK) * JSPLIT, 256, 0, stream>>>(xnh, rs);
  finalize_kernel<<<1, 256, 0, stream>>>(races, norms, dotab, dm, rs, out);
}
